// TensorProductHippocampus_9448928051669
// MI455X (gfx1250) — matrix operand order repaired and checked by reading, not compiled, not run on hardware
//
#include <hip/hip_runtime.h>

// ---------------------------------------------------------------------------
// Types
// ---------------------------------------------------------------------------
typedef __attribute__((ext_vector_type(16))) __bf16        v16bf;
typedef __attribute__((ext_vector_type(8)))  float         v8f;
typedef __attribute__((ext_vector_type(4)))  unsigned int  u32x4;
typedef __attribute__((ext_vector_type(4)))  int           i32x4;
typedef __attribute__((ext_vector_type(8)))  int           i32x8;

#define BB 64
#define TT 4096
#define DD 2048
#define RR 256
#define NCH 8
#define TCHUNK 512
#define TILES (TCHUNK / 16)   // 32 tiles of 16 rows per chunk

#if defined(__HIP_DEVICE_COMPILE__) && __has_builtin(__builtin_amdgcn_tensor_load_to_lds)
#define HAS_TDM 1
#else
#define HAS_TDM 0
#endif

// ---------------------------------------------------------------------------
// TDM helper: 2D tile load (dim0 f32 elements wide, `rows` rows, row stride
// `stride0` elements) from global to LDS. D# layout per CDNA5 ISA.
// ---------------------------------------------------------------------------
#if HAS_TDM
__device__ __forceinline__ void tdm_load_2d(const void* gptr, void* lptr,
                                            unsigned dim0, unsigned rows,
                                            unsigned stride0) {
  unsigned long long ga = (unsigned long long)gptr;
  unsigned lds = (unsigned)(unsigned long long)lptr;  // addr[31:0] = LDS offset
  u32x4 g0;
  g0[0] = 1u;                                          // count=1 (valid), user mode
  g0[1] = lds;                                         // lds_addr
  g0[2] = (unsigned)ga;                                // global_addr[31:0]
  g0[3] = (unsigned)((ga >> 32) & 0x1FFFFFFull) | 0x80000000u; // [56:32] | type=2
  i32x8 g1;
  g1[0] = (int)(2u << 16);                             // data_size=2 -> 4 bytes
  g1[1] = (int)((dim0 & 0xFFFFu) << 16);               // tensor_dim0[15:0]
  g1[2] = (int)((dim0 >> 16) | ((rows & 0xFFFFu) << 16)); // dim0 hi | tensor_dim1 lo
  g1[3] = (int)((rows >> 16) | (dim0 << 16));          // dim1 hi | tile_dim0 (=dim0)
  g1[4] = (int)(rows & 0xFFFFu);                       // tile_dim1 | tile_dim2=0
  g1[5] = (int)stride0;                                // tensor_dim0_stride lo
  g1[6] = 0;
  g1[7] = 0;
  i32x4 z4 = {0, 0, 0, 0};
#if __clang_major__ >= 23
  i32x8 z8 = {0, 0, 0, 0, 0, 0, 0, 0};
  __builtin_amdgcn_tensor_load_to_lds(g0, g1, z4, z4, z8, 0);
#else
  __builtin_amdgcn_tensor_load_to_lds(g0, g1, z4, z4, 0);
#endif
}
#endif

__device__ __forceinline__ unsigned short bf16_bits(float f) {
  return __builtin_bit_cast(unsigned short, (__bf16)f);
}

// ---------------------------------------------------------------------------
// Kernel 1: pre-swizzle W_k into bf16 WMMA B-fragment layout.
// Fragment (nt, ks): lane L holds n = nt*16 + L%16, K = ks*32 + 16*(L/16) + e,
// e=0..15, packed 2 bf16 per dword, 8 dwords (32B) per lane, contiguous.
// ---------------------------------------------------------------------------
__global__ __launch_bounds__(256) void wk_swizzle_kernel(
    const float* __restrict__ Wk, unsigned* __restrict__ wkz) {
  int g = blockIdx.x * 256 + threadIdx.x;  // 0..32767 dwords
  int frag = g >> 8;                       // 0..127  (nt*8 + ks)
  int r = g & 255;
  int lane = r >> 3;
  int j = r & 7;
  int nt = frag >> 3;
  int ks = frag & 7;
  int n = nt * 16 + (lane & 15);
  int hh = lane >> 4;
  int k0 = ks * 32 + 16 * hh + 2 * j;
  unsigned lo = bf16_bits(Wk[n * RR + k0]);
  unsigned hi = bf16_bits(Wk[n * RR + k0 + 1]);
  wkz[g] = lo | (hi << 16);
}

// ---------------------------------------------------------------------------
// Kernel 2: q = l2norm((wm @ Wc^T) @ Wq^T), one block per batch row.
// ---------------------------------------------------------------------------
__global__ __launch_bounds__(256) void q_kernel(
    const float* __restrict__ wm, const float* __restrict__ Wc,
    const float* __restrict__ Wq, float* __restrict__ qout) {
  __shared__ float wm_s[DD];
  __shared__ float wmc_s[RR];
  __shared__ float red_s[8];
  int b = blockIdx.x, tid = threadIdx.x;
  for (int i = 0; i < 8; ++i) wm_s[tid + 256 * i] = wm[(size_t)b * DD + tid + 256 * i];
  __syncthreads();
  const float* wr = Wc + (size_t)tid * DD;
  float c = 0.f;
  for (int d = 0; d < DD; d += 4)
    c += wr[d] * wm_s[d] + wr[d + 1] * wm_s[d + 1] + wr[d + 2] * wm_s[d + 2] +
         wr[d + 3] * wm_s[d + 3];
  wmc_s[tid] = c;
  __syncthreads();
  const float* qr = Wq + (size_t)tid * RR;
  float t = 0.f;
  for (int r = 0; r < RR; r += 4)
    t += qr[r] * wmc_s[r] + qr[r + 1] * wmc_s[r + 1] + qr[r + 2] * wmc_s[r + 2] +
         qr[r + 3] * wmc_s[r + 3];
  float ss = t * t;
  for (int m = 1; m < 32; m <<= 1) ss += __shfl_xor(ss, m);
  if ((tid & 31) == 0) red_s[tid >> 5] = ss;
  __syncthreads();
  float n2 = 0.f;
  for (int w = 0; w < 8; ++w) n2 += red_s[w];
  float nrm = fmaxf(sqrtf(n2), 1e-12f);
  qout[(size_t)b * RR + tid] = t / nrm;
}

// ---------------------------------------------------------------------------
// Kernel 3: streaming attention, online softmax + weighted tape row-sum.
// Grid: 64 batches x 8 T-chunks. 256 threads = 8 waves.
// Double-buffered 16x256 f32 tape tiles fetched by the Tensor Data Mover,
// prefetch issued before waiting (tensorcnt<=1) to overlap DMA with WMMA.
// Dynamic LDS: 128K swizzled bf16 W_k + 2x16K tape tiles + scratch = 166K.
// ---------------------------------------------------------------------------
#define SMEM_ATTN (131072 + 32768 + 1024 + 512 + 512 + 64 + 64)

__global__ __launch_bounds__(256) void attn_stream_kernel(
    const float* __restrict__ tape, const float* __restrict__ qv,
    const unsigned* __restrict__ wkz, const float* __restrict__ beta_p,
    float* __restrict__ chunkM, float* __restrict__ chunkS,
    float* __restrict__ chunkAcc) {
  extern __shared__ unsigned char smem_raw[];
  unsigned* wk_s  = (unsigned*)smem_raw;                        // 32768 dwords
  float* tape_s   = (float*)(smem_raw + 131072);                // 2 x 16 x 256
  float* q_s      = (float*)(smem_raw + 131072 + 32768);        // 256
  float* red_n2   = q_s + 256;                                  // 16 x 8
  float* red_qk   = red_n2 + 128;                               // 16 x 8
  float* score_s  = red_qk + 128;                               // 16
  float* mag_s    = score_s + 16;                               // 16

  const int tid = threadIdx.x;
  const int w = tid >> 5;
  const int lane = tid & 31;
  const int l16 = lane & 15;
  const int h = lane >> 4;
  const int b = blockIdx.x / NCH;
  const int ch = blockIdx.x % NCH;
  const int t0 = ch * TCHUNK;
  const float* tape_base = tape + ((size_t)b * TT + t0) * RR;

  // ---- prologue: stage swizzled W_k + first tape tile ----
#if HAS_TDM
  if (w == 0) {
    tdm_load_2d(wkz, wk_s, 32768u, 1u, 32768u);
    tdm_load_2d(tape_base, tape_s, 256u, 16u, 256u);
  }
#else
  for (int i = tid; i < 32768; i += 256) wk_s[i] = wkz[i];
  for (int i = tid; i < 16 * RR; i += 256) tape_s[i] = tape_base[i];
#endif
  q_s[tid] = qv[(size_t)b * RR + tid];
  __syncthreads();

  const float q0 = q_s[32 * w + l16];
  const float q1 = q_s[32 * w + 16 + l16];
  const float be = beta_p[0];
  const float sp = (be > 20.f) ? be : log1pf(__expf(be));

  float m = -3.0e38f, s = 0.f, acc = 0.f;

  for (int it = 0; it < TILES; ++it) {
    float* tbuf = tape_s + (it & 1) * (16 * RR);
    // prefetch next tile into the other buffer, then wait for current tile
#if HAS_TDM
    if (w == 0) {
      if (it + 1 < TILES) {
        tdm_load_2d(tape_base + (size_t)(it + 1) * 16 * RR,
                    tape_s + ((it + 1) & 1) * (16 * RR), 256u, 16u, 256u);
        __builtin_amdgcn_s_wait_tensorcnt(1);  // in-order: current tile done
      } else {
        __builtin_amdgcn_s_wait_tensorcnt(0);
      }
    }
#else
    if (it + 1 < TILES) {
      const float* src = tape_base + (size_t)(it + 1) * 16 * RR;
      float* dst = tape_s + ((it + 1) & 1) * (16 * RR);
      for (int i = tid; i < 16 * RR; i += 256) dst[i] = src[i];
    }
#endif
    __syncthreads();  // tile `it` resident in tbuf

    // |tape| row magnitudes: 16 threads/row, 16 contiguous floats each
    {
      int r = tid >> 4;
      const float* rp = tbuf + r * RR + (tid & 15) * 16;
      float4 a0 = *(const float4*)(rp);
      float4 a1 = *(const float4*)(rp + 4);
      float4 a2 = *(const float4*)(rp + 8);
      float4 a3 = *(const float4*)(rp + 12);
      float mg = fabsf(a0.x) + fabsf(a0.y) + fabsf(a0.z) + fabsf(a0.w) +
                 fabsf(a1.x) + fabsf(a1.y) + fabsf(a1.z) + fabsf(a1.w) +
                 fabsf(a2.x) + fabsf(a2.y) + fabsf(a2.z) + fabsf(a2.w) +
                 fabsf(a3.x) + fabsf(a3.y) + fabsf(a3.z) + fabsf(a3.w);
#pragma unroll
      for (int msk = 1; msk < 16; msk <<= 1) mg += __shfl_xor(mg, msk);
      if ((tid & 15) == 0) mag_s[r] = mg;
    }

    // A fragments: row M = l16, K runs ks*32+16h .. +15 (contiguous, matches B)
    v16bf A[8];
    {
      const float* rowp = tbuf + l16 * RR;
#pragma unroll
      for (int ks = 0; ks < 8; ++ks) {
        const float4* p1 = (const float4*)(rowp + ks * 32 + 16 * h);
        float4 r0 = p1[0], r1 = p1[1];          // K = ks*32+16h .. +7
        const float4* p2 = p1;
        float4 r2 = p2[2], r3 = p2[3];          // K = ks*32+16h+8 .. +15
        A[ks][0] = (__bf16)r0.x;  A[ks][1] = (__bf16)r0.y;
        A[ks][2] = (__bf16)r0.z;  A[ks][3] = (__bf16)r0.w;
        A[ks][4] = (__bf16)r1.x;  A[ks][5] = (__bf16)r1.y;
        A[ks][6] = (__bf16)r1.z;  A[ks][7] = (__bf16)r1.w;
        A[ks][8] = (__bf16)r2.x;  A[ks][9] = (__bf16)r2.y;
        A[ks][10] = (__bf16)r2.z; A[ks][11] = (__bf16)r2.w;
        A[ks][12] = (__bf16)r3.x; A[ks][13] = (__bf16)r3.y;
        A[ks][14] = (__bf16)r3.z; A[ks][15] = (__bf16)r3.w;
      }
    }

    float pn2[8], pqk[8];
#pragma unroll
    for (int c = 0; c < 8; ++c) { pn2[c] = 0.f; pqk[c] = 0.f; }

#pragma unroll
    for (int t2 = 0; t2 < 2; ++t2) {
      const int nt = 2 * w + t2;
      v8f C = {0.f, 0.f, 0.f, 0.f, 0.f, 0.f, 0.f, 0.f};
#pragma unroll
      for (int ks = 0; ks < 8; ++ks) {
        union { v16bf v; uint4 q[2]; } Bf;
        const uint4* bp = (const uint4*)(wk_s + ((nt * 8 + ks) * 32 + lane) * 8);
        Bf.q[0] = bp[0];
        Bf.q[1] = bp[1];
        C = __builtin_amdgcn_wmma_f32_16x16x32_bf16(false, A[ks], false, Bf.v,
                                                    (short)0, C, false, false);
      }
      const float qn = (t2 == 0) ? q0 : q1;
#pragma unroll
      for (int c = 0; c < 8; ++c) {
        float kv = C[c];
        pn2[c] += kv * kv;
        pqk[c] += qn * kv;
      }
    }

    // reduce over the 16 lanes of each half-wave (n dimension)
#pragma unroll
    for (int c = 0; c < 8; ++c) {
      float a2 = pn2[c], aq = pqk[c];
#pragma unroll
      for (int msk = 1; msk < 16; msk <<= 1) {
        a2 += __shfl_xor(a2, msk);
        aq += __shfl_xor(aq, msk);
      }
      if (l16 == 0) {
        red_n2[(c + 8 * h) * 8 + w] = a2;
        red_qk[(c + 8 * h) * 8 + w] = aq;
      }
    }
    __syncthreads();

    if (tid < 16) {
      float n2 = 0.f, qk = 0.f;
#pragma unroll
      for (int ww = 0; ww < 8; ++ww) {
        n2 += red_n2[tid * 8 + ww];
        qk += red_qk[tid * 8 + ww];
      }
      float nrm = fmaxf(sqrtf(n2), 1e-12f);
      float sc = sp * qk / nrm;
      if (mag_s[tid] < 1e-6f) sc = -1e9f;
      score_s[tid] = sc;
    }
    __syncthreads();

    // online softmax update (uniform across threads; acc per feature tid)
    {
      float scv[16];
      float tmax = -3.0e38f;
#pragma unroll
      for (int r = 0; r < 16; ++r) {
        scv[r] = score_s[r];
        tmax = fmaxf(tmax, scv[r]);
      }
      float mnew = fmaxf(m, tmax);
      float scale = __expf(m - mnew);
      float psum = 0.f, av = 0.f;
#pragma unroll
      for (int r = 0; r < 16; ++r) {
        float p = __expf(scv[r] - mnew);
        psum += p;
        av += p * tbuf[r * RR + tid];
      }
      s = s * scale + psum;
      acc = acc * scale + av;
      m = mnew;
    }
    __syncthreads();  // tile consumed: its buffer may be overwritten next iter
  }

  size_t ci = (size_t)(b * NCH + ch);
  chunkAcc[ci * RR + tid] = acc;
  if (tid == 0) { chunkM[ci] = m; chunkS[ci] = s; }
}

// ---------------------------------------------------------------------------
// Kernel 4: combine chunk softmax partials, apply W_v and W_decompress,
// emit retrieved (f32) and x = [wm, retrieved] as bf16 for the gate GEMM.
// ---------------------------------------------------------------------------
__global__ __launch_bounds__(256) void combine_kernel(
    const float* __restrict__ chunkM, const float* __restrict__ chunkS,
    const float* __restrict__ chunkAcc, const float* __restrict__ Wv,
    const float* __restrict__ Wd, const float* __restrict__ wm,
    float* __restrict__ retrieved, unsigned short* __restrict__ xb) {
  __shared__ float wsum_s[RR];
  __shared__ float rc_s[RR];
  int b = blockIdx.x, tid = threadIdx.x;
  float M = -3.0e38f;
  for (int c = 0; c < NCH; ++c) M = fmaxf(M, chunkM[b * NCH + c]);
  float S = 0.f, wsum = 0.f;
  for (int c = 0; c < NCH; ++c) {
    float e = __expf(chunkM[b * NCH + c] - M);
    S += chunkS[b * NCH + c] * e;
    wsum += chunkAcc[(size_t)(b * NCH + c) * RR + tid] * e;
  }
  wsum_s[tid] = (S > 0.f) ? wsum / S : 0.f;
  __syncthreads();
  const float* vr = Wv + (size_t)tid * RR;
  float rc = 0.f;
  for (int r = 0; r < RR; r += 4)
    rc += vr[r] * wsum_s[r] + vr[r + 1] * wsum_s[r + 1] +
          vr[r + 2] * wsum_s[r + 2] + vr[r + 3] * wsum_s[r + 3];
  rc_s[tid] = rc;
  __syncthreads();
  for (int i = 0; i < 8; ++i) {
    int d = i * 256 + tid;
    const float* wd = Wd + (size_t)d * RR;
    float rv = 0.f;
    for (int r = 0; r < RR; r += 4)
      rv += wd[r] * rc_s[r] + wd[r + 1] * rc_s[r + 1] + wd[r + 2] * rc_s[r + 2] +
            wd[r + 3] * rc_s[r + 3];
    retrieved[(size_t)b * DD + d] = rv;
    xb[(size_t)b * (2 * DD) + DD + d] = bf16_bits(rv);
    xb[(size_t)b * (2 * DD) + d] = bf16_bits(wm[(size_t)b * DD + d]);
  }
}

// ---------------------------------------------------------------------------
// Kernel 5: gate pre-activation  g = x[64,4096] @ W_gate^T[4096,2048]
// bf16 WMMA, deterministic split-K into 4 partial buffers. 1 wave per block.
// ---------------------------------------------------------------------------
__global__ __launch_bounds__(32) void gate_wmma_kernel(
    const unsigned short* __restrict__ xb, const float* __restrict__ Wg,
    float* __restrict__ gate_part) {
  const int nt = blockIdx.x;   // 0..127 column tiles of 16
  const int ky = blockIdx.y;   // 0..3 K splits (1024 each)
  const int lane = threadIdx.x;
  const int l16 = lane & 15;
  const int h = lane >> 4;
  const int n = nt * 16 + l16;

  v8f C[4];
#pragma unroll
  for (int mt = 0; mt < 4; ++mt) {
    v8f z = {0.f, 0.f, 0.f, 0.f, 0.f, 0.f, 0.f, 0.f};
    C[mt] = z;
  }

  for (int ks = 0; ks < 32; ++ks) {
    const int kk = ky * 32 + ks;  // global K step, K offset = kk*32
    v16bf Bf;
    const float4* wp = (const float4*)(Wg + (size_t)n * (2 * DD) + kk * 32 + 16 * h);
    float4 w0 = wp[0], w1 = wp[1], w2 = wp[2], w3 = wp[3];
    Bf[0] = (__bf16)w0.x;  Bf[1] = (__bf16)w0.y;
    Bf[2] = (__bf16)w0.z;  Bf[3] = (__bf16)w0.w;
    Bf[4] = (__bf16)w1.x;  Bf[5] = (__bf16)w1.y;
    Bf[6] = (__bf16)w1.z;  Bf[7] = (__bf16)w1.w;
    Bf[8] = (__bf16)w2.x;  Bf[9] = (__bf16)w2.y;
    Bf[10] = (__bf16)w2.z; Bf[11] = (__bf16)w2.w;
    Bf[12] = (__bf16)w3.x; Bf[13] = (__bf16)w3.y;
    Bf[14] = (__bf16)w3.z; Bf[15] = (__bf16)w3.w;
#pragma unroll
    for (int mt = 0; mt < 4; ++mt) {
      const int row = mt * 16 + l16;
      const unsigned short* xp = xb + (size_t)row * (2 * DD) + kk * 32 + 16 * h;
      union { v16bf v; uint4 q[2]; } Af;
      Af.q[0] = *(const uint4*)(xp);
      Af.q[1] = *(const uint4*)(xp + 8);
      C[mt] = __builtin_amdgcn_wmma_f32_16x16x32_bf16(false, Af.v, false, Bf,
                                                      (short)0, C[mt], false,
                                                      false);
    }
  }

  float* gp = gate_part + (size_t)ky * (BB * DD);
#pragma unroll
  for (int mt = 0; mt < 4; ++mt)
#pragma unroll
    for (int c = 0; c < 8; ++c) {
      int row = mt * 16 + c + 8 * h;
      gp[(size_t)row * DD + nt * 16 + l16] = C[mt][c];
    }
}

// ---------------------------------------------------------------------------
// Kernel 6: out = wm + sigmoid(sum_k gate_part + b_gate) * retrieved
// ---------------------------------------------------------------------------
__global__ __launch_bounds__(256) void finalize_kernel(
    const float* __restrict__ wm, const float* __restrict__ retrieved,
    const float* __restrict__ gate_part, const float* __restrict__ bg,
    float* __restrict__ out) {
  int i = blockIdx.x * 256 + threadIdx.x;
  if (i >= BB * DD) return;
  int d = i & (DD - 1);
  float g = bg[d];
#pragma unroll
  for (int ky = 0; ky < 4; ++ky) g += gate_part[(size_t)ky * (BB * DD) + i];
  float sig = 1.f / (1.f + __expf(-g));
  out[i] = wm[i] + sig * retrieved[i];
}

// ---------------------------------------------------------------------------
// Host launcher
// ---------------------------------------------------------------------------
extern "C" void kernel_launch(void* const* d_in, const int* in_sizes, int n_in,
                              void* d_out, int out_size, void* d_ws,
                              size_t ws_size, hipStream_t stream) {
  const float* wm   = (const float*)d_in[0];
  const float* tape = (const float*)d_in[1];
  const float* Wc   = (const float*)d_in[2];
  const float* Wd   = (const float*)d_in[3];
  const float* Wq   = (const float*)d_in[4];
  const float* Wk   = (const float*)d_in[5];
  const float* Wv   = (const float*)d_in[6];
  const float* Wg   = (const float*)d_in[7];
  const float* bg   = (const float*)d_in[8];
  const float* beta = (const float*)d_in[9];

  char* ws = (char*)d_ws;
  float*          q_buf     = (float*)(ws + 0);                //  64K
  unsigned*       wkz       = (unsigned*)(ws + 65536);         // 128K
  float*          chunkM    = (float*)(ws + 196608);           //   2K
  float*          chunkS    = (float*)(ws + 198656);           //   2K
  float*          chunkAcc  = (float*)(ws + 200704);           // 512K
  float*          retrieved = (float*)(ws + 724992);           // 512K
  unsigned short* xb        = (unsigned short*)(ws + 1249280); // 512K
  float*          gate_part = (float*)(ws + 1773568);          //   2M

  wk_swizzle_kernel<<<128, 256, 0, stream>>>(Wk, wkz);
  q_kernel<<<64, 256, 0, stream>>>(wm, Wc, Wq, q_buf);
  attn_stream_kernel<<<BB * NCH, 256, SMEM_ATTN, stream>>>(
      tape, q_buf, wkz, beta, chunkM, chunkS, chunkAcc);
  combine_kernel<<<64, 256, 0, stream>>>(chunkM, chunkS, chunkAcc, Wv, Wd, wm,
                                         retrieved, xb);
  dim3 gridD(128, 4);
  gate_wmma_kernel<<<gridD, 32, 0, stream>>>(xb, Wg, gate_part);
  finalize_kernel<<<(BB * DD + 255) / 256, 256, 0, stream>>>(
      wm, retrieved, gate_part, bg, (float*)d_out);
}
